// MultiHeadAttention_18468359373520
// MI455X (gfx1250) — compile-verified, matrix operand order checked by reading, not run on hardware
//
#include <hip/hip_runtime.h>
#include <hip/hip_bf16.h>
#include <stdint.h>

#ifndef __has_builtin
#define __has_builtin(x) 0
#endif
#if __has_builtin(__builtin_amdgcn_tensor_load_to_lds)
#define USE_TDM 1
#else
#define USE_TDM 0
#endif

#define D_MODEL 1024
#define H_HEADS 16
#define DK      64
#define BATCH   4
#define SEQ     2048
#define M_TOT   (BATCH * SEQ)   // 8192

typedef __attribute__((ext_vector_type(16))) __bf16 v16bf;
typedef __attribute__((ext_vector_type(8)))  float  v8f;
typedef __attribute__((ext_vector_type(4)))  unsigned int v4u;

union FragA {
    v16bf v;
    __bf16 h[16];
    v4u   q[2];
};

__device__ __forceinline__ __bf16 f2bf(float f) {
    union { float f; uint32_t u; } c; c.f = f;
    uint32_t lsb = (c.u >> 16) & 1u;
    c.u += 0x7fffu + lsb;                  // round-to-nearest-even
    union { uint16_t s; __bf16 b; } o; o.s = (uint16_t)(c.u >> 16);
    return o.b;
}

#if USE_TDM
typedef __attribute__((ext_vector_type(8))) int v8i;
typedef __attribute__((ext_vector_type(4))) int v4i;

// Build a 2D Tensor-DMA descriptor (D# groups 0/1) and issue TENSOR_LOAD_TO_LDS.
// Layout per CDNA5 ISA §8.3/8.4: count=1, type=2 (image), data_size=1 (2 bytes),
// tensor_dim0/1, tile_dim0/1, tensor_dim0_stride; groups 2/3 zero (2D tensor).
__device__ __forceinline__ void tdm_load_2d(unsigned lds_addr, const __bf16* gptr,
                                            unsigned tensor_d0, unsigned tensor_d1,
                                            unsigned stride_d0,
                                            unsigned tile_d0, unsigned tile_d1) {
    unsigned long long ga = (unsigned long long)(uintptr_t)gptr;
    v4u g0;
    g0[0] = 1u;                                            // count=1, user desc
    g0[1] = lds_addr;                                      // LDS byte address
    g0[2] = (unsigned)ga;                                  // global addr [31:0]
    g0[3] = (unsigned)((ga >> 32) & 0x01FFFFFFull) | (2u << 30);  // [56:32] | type=2
    v8i g1;
    g1[0] = (int)(1u << 16);                               // data_size = 2 bytes
    g1[1] = (int)((tensor_d0 & 0xFFFFu) << 16);            // tensor_dim0 bits 79:48
    g1[2] = (int)(((tensor_d0 >> 16) & 0xFFFFu) |
                  ((tensor_d1 & 0xFFFFu) << 16));          // tensor_dim1 bits 111:80
    g1[3] = (int)(((tensor_d1 >> 16) & 0xFFFFu) |
                  ((tile_d0 & 0xFFFFu) << 16));            // tile_dim0 bits 127:112
    g1[4] = (int)(tile_d1 & 0xFFFFu);                      // tile_dim1; tile_dim2=0
    g1[5] = (int)stride_d0;                                // tensor_dim0_stride
    g1[6] = 0;
    g1[7] = 0;
    v4i gz = (v4i){0, 0, 0, 0};
#if __clang_major__ >= 23
    v8i gz8 = (v8i){0, 0, 0, 0, 0, 0, 0, 0};
    __builtin_amdgcn_tensor_load_to_lds(g0, g1, gz, gz, gz8, 0);
#else
    __builtin_amdgcn_tensor_load_to_lds(g0, g1, gz, gz, 0);
#endif
}
#endif  // USE_TDM

// ---------------------------------------------------------------------------
// Kernel 1: fp32 -> bf16 cast (vectorized)
// ---------------------------------------------------------------------------
__global__ void cast_f32_bf16(const float* __restrict__ src,
                              __bf16* __restrict__ dst, int n) {
    int i = (blockIdx.x * blockDim.x + threadIdx.x) * 4;
    if (i + 3 < n) {
        float4 f = *(const float4*)(src + i);
        dst[i + 0] = f2bf(f.x);
        dst[i + 1] = f2bf(f.y);
        dst[i + 2] = f2bf(f.z);
        dst[i + 3] = f2bf(f.w);
    } else {
        for (int j = i; j < n; ++j) dst[j] = f2bf(src[j]);
    }
}

// ---------------------------------------------------------------------------
// Kernel 1b: fp32 [H][1024][64] -> bf16 transposed [H][64][1024]
// ---------------------------------------------------------------------------
__global__ void cast_transpose_w(const float* __restrict__ src,
                                 __bf16* __restrict__ dst) {
    int idx = blockIdx.x * blockDim.x + threadIdx.x;   // over H*1024*64
    int h   = idx >> 16;
    int rem = idx & 0xFFFF;
    int kk  = rem >> 6;
    int n   = rem & 63;
    dst[((size_t)h << 16) + (size_t)n * D_MODEL + kk] = f2bf(src[idx]);
}

// ---------------------------------------------------------------------------
// Kernel 2: per-head projection GEMM, double-buffered LDS.
//   X  : [8192][1024] bf16
//   Wt : [H][64][1024] bf16   (pre-transposed -> column-major B)
//   Y  : vtrans==0 : [H][8192][64] bf16
//        vtrans==1 : [H][B][64][SEQ] bf16   (V stored transposed for attn)
// Block: 256 threads = 8 waves; tile M=128, N=64, K-step 64; 1 barrier/step.
// ---------------------------------------------------------------------------
__global__ __launch_bounds__(256)
void proj_gemm(const __bf16* __restrict__ X,
               const __bf16* __restrict__ Wt,
               __bf16* __restrict__ Y,
               int vtrans) {
    __shared__ __bf16 As[2][128][64];   // 32 KB
    __shared__ __bf16 Bt[2][64][64];    // 16 KB
    const int h    = blockIdx.y;
    const int m0   = blockIdx.x * 128;
    const int tid  = threadIdx.x;
    const int w    = tid >> 5;
    const int lane = tid & 31;
    const int l16  = lane & 15;
    const int half = lane >> 4;

    v8f acc[4];
    #pragma unroll
    for (int nt = 0; nt < 4; ++nt) acc[nt] = (v8f){0,0,0,0,0,0,0,0};

    const __bf16* Wh = Wt + ((size_t)h << 16);   // h * 64 * 1024

    v4u areg[4], breg[2];
    auto gload = [&](int k0) {
        #pragma unroll
        for (int it = 0; it < 4; ++it) {
            int idx = tid + it * 256;
            int r = idx >> 3, c8 = idx & 7;
            areg[it] = *(const v4u*)(X + (size_t)(m0 + r) * D_MODEL + k0 + c8 * 8);
        }
        #pragma unroll
        for (int it = 0; it < 2; ++it) {
            int idx = tid + it * 256;
            int n = idx >> 3, c8 = idx & 7;
            breg[it] = *(const v4u*)(Wh + (size_t)n * D_MODEL + k0 + c8 * 8);
        }
    };
    auto lstore = [&](int buf) {
        #pragma unroll
        for (int it = 0; it < 4; ++it) {
            int idx = tid + it * 256;
            int r = idx >> 3, c8 = idx & 7;
            *(v4u*)(&As[buf][r][c8 * 8]) = areg[it];
        }
        #pragma unroll
        for (int it = 0; it < 2; ++it) {
            int idx = tid + it * 256;
            int n = idx >> 3, c8 = idx & 7;
            *(v4u*)(&Bt[buf][n][c8 * 8]) = breg[it];
        }
    };

    const int NIT = D_MODEL / 64;    // 16
    gload(0);
    lstore(0);
    for (int i = 0; i < NIT; ++i) {
        __syncthreads();
        if (i + 1 < NIT) gload((i + 1) * 64);
        const int buf = i & 1;
        #pragma unroll
        for (int kc = 0; kc < 2; ++kc) {
            FragA af;
            int row = w * 16 + l16;
            af.q[0] = *(const v4u*)(&As[buf][row][kc * 32 + half * 8]);
            af.q[1] = *(const v4u*)(&As[buf][row][kc * 32 + 16 + half * 8]);
            FragA bfr[4];
            #pragma unroll
            for (int nt = 0; nt < 4; ++nt) {
                int n = nt * 16 + l16;
                bfr[nt].q[0] = *(const v4u*)(&Bt[buf][n][kc * 32 + half * 8]);
                bfr[nt].q[1] = *(const v4u*)(&Bt[buf][n][kc * 32 + 16 + half * 8]);
            }
            #pragma unroll
            for (int nt = 0; nt < 4; ++nt)
                acc[nt] = __builtin_amdgcn_wmma_f32_16x16x32_bf16(
                    false, af.v, false, bfr[nt].v, (short)0, acc[nt], false, false);
        }
        if (i + 1 < NIT) lstore((i + 1) & 1);
    }

    if (vtrans == 0) {
        __bf16* Yh = Y + (size_t)h * ((size_t)M_TOT * DK);
        #pragma unroll
        for (int nt = 0; nt < 4; ++nt)
            #pragma unroll
            for (int r = 0; r < 8; ++r) {
                int m = m0 + w * 16 + r + 8 * half;     // C-layout row striping
                Yh[(size_t)m * DK + nt * 16 + l16] = f2bf(acc[nt][r]);
            }
    } else {
        // V^T layout: [(h*B + b)*DK + n][s]
        #pragma unroll
        for (int nt = 0; nt < 4; ++nt)
            #pragma unroll
            for (int r = 0; r < 8; ++r) {
                int m = m0 + w * 16 + r + 8 * half;
                int b = m >> 11, s = m & (SEQ - 1);
                int n = nt * 16 + l16;
                Y[(((size_t)h * BATCH + b) * DK + n) * SEQ + s] = f2bf(acc[nt][r]);
            }
    }
}

// ---------------------------------------------------------------------------
// Kernel 3: flash attention (online softmax), double-buffered K/V tiles.
// K/V tiles staged via Tensor Data Mover (TDM) when available, issued by
// wave 0 only (uniform branch), completion via s_wait_tensorcnt.
//   Qh/Kh : [H][B*S][64] bf16
//   VhT   : [H][B][64][SEQ] bf16
//   AO    : [B*S][1024] bf16 (head-major feature concat)
// Block: 128 threads = 4 waves; 64 queries per block; 64-key tiles.
// ---------------------------------------------------------------------------
__global__ __launch_bounds__(128)
void attn_kernel(const __bf16* __restrict__ Qh,
                 const __bf16* __restrict__ Kh,
                 const __bf16* __restrict__ VhT,
                 __bf16* __restrict__ AO) {
    __shared__ __bf16 Kt[2][64][64];    // keys row-major   (16 KB)
    __shared__ __bf16 Vt[2][64][64];    // [dk][key]        (16 KB)
    __shared__ __bf16 Pt[4][16][64];    // wave-private P relayout (8 KB)

    const int tid  = threadIdx.x;
    const int w    = tid >> 5;
    const int lane = tid & 31;
    const int l16  = lane & 15;
    const int half = lane >> 4;
    const int hb   = blockIdx.y;
    const int b    = hb / H_HEADS;
    const int h    = hb % H_HEADS;
    const int q0   = blockIdx.x * 64;
    const size_t headBase = ((size_t)h * M_TOT + (size_t)b * SEQ) * DK;
    const __bf16* vbase = VhT + ((size_t)h * BATCH + b) * DK * SEQ;
    const float scale = 0.125f;       // 1/sqrt(64)

    // Preload Q fragments for this wave's 16 rows (dk = 64 -> 2 frags)
    FragA aq[2];
    {
        const __bf16* qp = Qh + headBase + (size_t)(q0 + w * 16 + l16) * DK;
        #pragma unroll
        for (int kc = 0; kc < 2; ++kc) {
            aq[kc].q[0] = *(const v4u*)(qp + kc * 32 + half * 8);
            aq[kc].q[1] = *(const v4u*)(qp + kc * 32 + 16 + half * 8);
        }
    }

    v8f o[4];
    #pragma unroll
    for (int nt = 0; nt < 4; ++nt) o[nt] = (v8f){0,0,0,0,0,0,0,0};
    float mrow[8], lrow[8];
    #pragma unroll
    for (int r = 0; r < 8; ++r) { mrow[r] = -3.0e38f; lrow[r] = 0.f; }

#if USE_TDM
    // TDM issue wave: must be a scalar-uniform decision (TDM ignores EXEC).
    const bool tdmw = ((__builtin_amdgcn_readfirstlane((int)tid) >> 5) == 0);
    auto issue_tile = [&](int kt, int buf) {
        // K tile: rows kt..kt+63 of [SEQ][DK]  ->  Kt[buf] (contiguous)
        tdm_load_2d((unsigned)(uintptr_t)&Kt[buf][0][0],
                    Kh + headBase + (size_t)kt * DK,
                    /*tensor_d0=*/DK, /*tensor_d1=*/SEQ, /*stride_d0=*/DK,
                    /*tile_d0=*/DK, /*tile_d1=*/64);
        // V tile: cols kt..kt+63 of [DK][SEQ]  ->  Vt[buf] (64 rows of 64)
        tdm_load_2d((unsigned)(uintptr_t)&Vt[buf][0][0],
                    vbase + kt,
                    /*tensor_d0=*/SEQ, /*tensor_d1=*/DK, /*stride_d0=*/SEQ,
                    /*tile_d0=*/64, /*tile_d1=*/DK);
    };
#else
    v4u kreg[4], vreg[4];
    auto gload = [&](int kt) {
        #pragma unroll
        for (int it = 0; it < 4; ++it) {
            int idx = tid + it * 128;
            int r = idx >> 3, c8 = idx & 7;
            kreg[it] = *(const v4u*)(Kh + headBase + (size_t)(kt + r) * DK + c8 * 8);
            vreg[it] = *(const v4u*)(vbase + (size_t)r * SEQ + kt + c8 * 8);
        }
    };
    auto lstore = [&](int buf) {
        #pragma unroll
        for (int it = 0; it < 4; ++it) {
            int idx = tid + it * 128;
            int r = idx >> 3, c8 = idx & 7;
            *(v4u*)(&Kt[buf][r][c8 * 8]) = kreg[it];
            *(v4u*)(&Vt[buf][r][c8 * 8]) = vreg[it];
        }
    };
#endif

    const int NT = SEQ / 64;    // 32 key tiles
#if USE_TDM
    if (tdmw) issue_tile(0, 0);
#else
    gload(0);
    lstore(0);
#endif

    for (int i = 0; i < NT; ++i) {
        const int buf = i & 1;
#if USE_TDM
        if (tdmw) __builtin_amdgcn_s_wait_tensorcnt(0);   // tile i landed in LDS
        __syncthreads();
        if (i + 1 < NT && tdmw) issue_tile((i + 1) * 64, buf ^ 1);
#else
        __syncthreads();
        if (i + 1 < NT) gload((i + 1) * 64);
#endif

        // Scores: four 16-key subtiles, contraction over dk=64 (2 WMMA each)
        float p[4][8];
        float newm[8];
        #pragma unroll
        for (int r = 0; r < 8; ++r) newm[r] = mrow[r];

        #pragma unroll
        for (int st = 0; st < 4; ++st) {
            v8f c = (v8f){0,0,0,0,0,0,0,0};
            #pragma unroll
            for (int kc = 0; kc < 2; ++kc) {
                FragA bk;
                int key = st * 16 + l16;
                bk.q[0] = *(const v4u*)(&Kt[buf][key][kc * 32 + half * 8]);
                bk.q[1] = *(const v4u*)(&Kt[buf][key][kc * 32 + 16 + half * 8]);
                c = __builtin_amdgcn_wmma_f32_16x16x32_bf16(
                    false, aq[kc].v, false, bk.v, (short)0, c, false, false);
            }
            #pragma unroll
            for (int r = 0; r < 8; ++r) {
                float s = c[r] * scale;
                p[st][r] = s;
                float tm = s;
                #pragma unroll
                for (int off = 1; off < 16; off <<= 1)   // stays inside 16-lane half
                    tm = fmaxf(tm, __shfl_xor(tm, off, 32));
                newm[r] = fmaxf(newm[r], tm);
            }
        }

        // Online-softmax update
        #pragma unroll
        for (int r = 0; r < 8; ++r) {
            float alpha = __expf(mrow[r] - newm[r]);
            mrow[r] = newm[r];
            float ps = 0.f;
            #pragma unroll
            for (int st = 0; st < 4; ++st) {
                float e = __expf(p[st][r] - newm[r]);
                p[st][r] = e;
                ps += e;
            }
            #pragma unroll
            for (int off = 1; off < 16; off <<= 1)
                ps += __shfl_xor(ps, off, 32);
            lrow[r] = lrow[r] * alpha + ps;
            #pragma unroll
            for (int nt = 0; nt < 4; ++nt) o[nt][r] *= alpha;
        }

        // Relayout P (C layout -> A-fragment layout) through wave-private LDS.
        // Same-wave DS ops complete in order: no workgroup barrier needed.
        #pragma unroll
        for (int st = 0; st < 4; ++st)
            #pragma unroll
            for (int r = 0; r < 8; ++r)
                Pt[w][r + 8 * half][st * 16 + l16] = f2bf(p[st][r]);

        FragA ap[2];
        #pragma unroll
        for (int kc = 0; kc < 2; ++kc) {
            ap[kc].q[0] = *(const v4u*)(&Pt[w][l16][kc * 32 + half * 8]);
            ap[kc].q[1] = *(const v4u*)(&Pt[w][l16][kc * 32 + 16 + half * 8]);
        }
        #pragma unroll
        for (int nt = 0; nt < 4; ++nt) {
            #pragma unroll
            for (int kc = 0; kc < 2; ++kc) {
                FragA bv;
                int nd = nt * 16 + l16;
                bv.q[0] = *(const v4u*)(&Vt[buf][nd][kc * 32 + half * 8]);
                bv.q[1] = *(const v4u*)(&Vt[buf][nd][kc * 32 + 16 + half * 8]);
                o[nt] = __builtin_amdgcn_wmma_f32_16x16x32_bf16(
                    false, ap[kc].v, false, bv.v, (short)0, o[nt], false, false);
            }
        }

#if !USE_TDM
        if (i + 1 < NT) lstore(buf ^ 1);
#endif
    }

    // Epilogue: normalize and write head-major concat [B*S][H*DK]
    #pragma unroll
    for (int r = 0; r < 8; ++r) {
        float rl = lrow[r] > 0.f ? 1.0f / lrow[r] : 0.f;
        int srow = q0 + w * 16 + r + 8 * half;
        size_t obase = ((size_t)b * SEQ + srow) * (size_t)D_MODEL + (size_t)h * DK;
        #pragma unroll
        for (int nt = 0; nt < 4; ++nt)
            AO[obase + nt * 16 + l16] = f2bf(o[nt][r] * rl);
    }
}

// ---------------------------------------------------------------------------
// Kernel 4: output projection  Y = X @ Wo^T + b, double-buffered LDS.
//   X  : [8192][1024] bf16
//   Wo : [1024][1024] bf16 (row = output feature) -> rows are column-major B
//   Y  : [8192][1024] f32
// ---------------------------------------------------------------------------
__global__ __launch_bounds__(256)
void out_proj(const __bf16* __restrict__ X,
              const __bf16* __restrict__ Wo,
              const float* __restrict__ bias,
              float* __restrict__ Y) {
    __shared__ __bf16 As[2][128][64];   // 32 KB
    __shared__ __bf16 Bt[2][64][64];    // 16 KB
    const int m0   = blockIdx.x * 128;
    const int n0   = blockIdx.y * 64;
    const int tid  = threadIdx.x;
    const int w    = tid >> 5;
    const int lane = tid & 31;
    const int l16  = lane & 15;
    const int half = lane >> 4;

    v8f acc[4];
    #pragma unroll
    for (int nt = 0; nt < 4; ++nt) acc[nt] = (v8f){0,0,0,0,0,0,0,0};

    v4u areg[4], breg[2];
    auto gload = [&](int k0) {
        #pragma unroll
        for (int it = 0; it < 4; ++it) {
            int idx = tid + it * 256;
            int r = idx >> 3, c8 = idx & 7;
            areg[it] = *(const v4u*)(X + (size_t)(m0 + r) * D_MODEL + k0 + c8 * 8);
        }
        #pragma unroll
        for (int it = 0; it < 2; ++it) {
            int idx = tid + it * 256;
            int n = idx >> 3, c8 = idx & 7;
            breg[it] = *(const v4u*)(Wo + (size_t)(n0 + n) * D_MODEL + k0 + c8 * 8);
        }
    };
    auto lstore = [&](int buf) {
        #pragma unroll
        for (int it = 0; it < 4; ++it) {
            int idx = tid + it * 256;
            int r = idx >> 3, c8 = idx & 7;
            *(v4u*)(&As[buf][r][c8 * 8]) = areg[it];
        }
        #pragma unroll
        for (int it = 0; it < 2; ++it) {
            int idx = tid + it * 256;
            int n = idx >> 3, c8 = idx & 7;
            *(v4u*)(&Bt[buf][n][c8 * 8]) = breg[it];
        }
    };

    const int NIT = D_MODEL / 64;    // 16
    gload(0);
    lstore(0);
    for (int i = 0; i < NIT; ++i) {
        __syncthreads();
        if (i + 1 < NIT) gload((i + 1) * 64);
        const int buf = i & 1;
        #pragma unroll
        for (int kc = 0; kc < 2; ++kc) {
            FragA af;
            int row = w * 16 + l16;
            af.q[0] = *(const v4u*)(&As[buf][row][kc * 32 + half * 8]);
            af.q[1] = *(const v4u*)(&As[buf][row][kc * 32 + 16 + half * 8]);
            FragA bfr[4];
            #pragma unroll
            for (int nt = 0; nt < 4; ++nt) {
                int n = nt * 16 + l16;
                bfr[nt].q[0] = *(const v4u*)(&Bt[buf][n][kc * 32 + half * 8]);
                bfr[nt].q[1] = *(const v4u*)(&Bt[buf][n][kc * 32 + 16 + half * 8]);
            }
            #pragma unroll
            for (int nt = 0; nt < 4; ++nt)
                acc[nt] = __builtin_amdgcn_wmma_f32_16x16x32_bf16(
                    false, af.v, false, bfr[nt].v, (short)0, acc[nt], false, false);
        }
        if (i + 1 < NIT) lstore((i + 1) & 1);
    }

    #pragma unroll
    for (int nt = 0; nt < 4; ++nt)
        #pragma unroll
        for (int r = 0; r < 8; ++r) {
            int m = m0 + w * 16 + r + 8 * half;
            int n = n0 + nt * 16 + l16;
            Y[(size_t)m * D_MODEL + n] = acc[nt][r] + bias[n];
        }
}

// ---------------------------------------------------------------------------
// Host launcher
// ---------------------------------------------------------------------------
extern "C" void kernel_launch(void* const* d_in, const int* in_sizes, int n_in,
                              void* d_out, int out_size, void* d_ws, size_t ws_size,
                              hipStream_t stream) {
    (void)in_sizes; (void)n_in; (void)out_size; (void)ws_size;
    const float* q   = (const float*)d_in[0];
    const float* k   = (const float*)d_in[1];
    const float* v   = (const float*)d_in[2];
    const float* W_q = (const float*)d_in[3];
    const float* W_k = (const float*)d_in[4];
    const float* W_v = (const float*)d_in[5];
    const float* W_o = (const float*)d_in[6];
    const float* b_o = (const float*)d_in[7];
    float* out = (float*)d_out;

    const size_t nAct = (size_t)M_TOT * D_MODEL;          // 8,388,608
    const size_t nW   = (size_t)H_HEADS * D_MODEL * DK;   // 1,048,576
    const size_t nWo  = (size_t)D_MODEL * D_MODEL;        // 1,048,576
    const size_t nH   = (size_t)H_HEADS * M_TOT * DK;     // 8,388,608

    char* ws = (char*)d_ws;
    size_t off = 0;
    auto carve = [&](size_t elems) -> __bf16* {
        __bf16* p = (__bf16*)(ws + off);
        off += ((elems * sizeof(__bf16) + 255) / 256) * 256;
        return p;
    };
    __bf16* qb  = carve(nAct);
    __bf16* kb  = carve(nAct);
    __bf16* vb  = carve(nAct);
    __bf16* wqt = carve(nW);     // [H][64][1024]
    __bf16* wkt = carve(nW);
    __bf16* wvt = carve(nW);
    __bf16* wob = carve(nWo);
    __bf16* qh  = carve(nH);     // [H][8192][64]
    __bf16* kh  = carve(nH);
    __bf16* vhT = carve(nH);     // [H][B][64][SEQ]
    __bf16* ao  = carve(nAct);

    auto cgrid = [](size_t n) { return (unsigned)((n / 4 + 255) / 256); };
    cast_f32_bf16<<<cgrid(nAct), 256, 0, stream>>>(q,   qb,  (int)nAct);
    cast_f32_bf16<<<cgrid(nAct), 256, 0, stream>>>(k,   kb,  (int)nAct);
    cast_f32_bf16<<<cgrid(nAct), 256, 0, stream>>>(v,   vb,  (int)nAct);
    cast_f32_bf16<<<cgrid(nWo),  256, 0, stream>>>(W_o, wob, (int)nWo);

    unsigned tw = (unsigned)(nW / 256);
    cast_transpose_w<<<tw, 256, 0, stream>>>(W_q, wqt);
    cast_transpose_w<<<tw, 256, 0, stream>>>(W_k, wkt);
    cast_transpose_w<<<tw, 256, 0, stream>>>(W_v, wvt);

    dim3 pg(M_TOT / 128, H_HEADS);                 // (64, 16)
    proj_gemm<<<pg, 256, 0, stream>>>(qb, wqt, qh,  0);
    proj_gemm<<<pg, 256, 0, stream>>>(kb, wkt, kh,  0);
    proj_gemm<<<pg, 256, 0, stream>>>(vb, wvt, vhT, 1);

    dim3 ag(SEQ / 64, H_HEADS * BATCH);            // (32, 64)
    attn_kernel<<<ag, 128, 0, stream>>>(qh, kh, vhT, ao);

    dim3 og(M_TOT / 128, D_MODEL / 64);            // (64, 16)
    out_proj<<<og, 256, 0, stream>>>(ao, wob, b_o, out);
}
